// Model_84464826843698
// MI455X (gfx1250) — matrix operand order repaired and checked by reading, not compiled, not run on hardware
//
#include <hip/hip_runtime.h>

typedef __attribute__((ext_vector_type(16))) __bf16 v16bf;
typedef __attribute__((ext_vector_type(8)))  float  v8f;

union BF16x16 { v16bf v; __bf16 e[16]; uint4 u[2]; };
union BF16x8  { __bf16 e[8]; uint4 u; };
union F32x8   { v8f v; float e[8]; };

#define CIN   128
#define NPTS  16384
#define KD    16
#define MIDC  64
#define OUTC  128
#define PTC   10
#define GROUP 16
#define ROWSTRIDE (NPTS*KD)   /* 262144 */
#define AR_STR 136            /* arena col stride, bf16 elems (272B, 16B aligned, bank-skewed) */
#define PP_STR 72             /* ppfs/pt col stride  (144B) */
#define XFS_STR 1032          /* xfs per-point stride (2064B) */

#define OFF_ARENA 0
#define OFF_PP    69632
#define OFF_XFS   106496
#define OFF_X1    139520
#define OFF_H     143616
#define OFF_PART  144128
#define OFF_WSM   152320
#define OFF_OM    160512
#define OFF_ID    162816
#define SMEM_TOTAL 171008

__device__ __forceinline__ v8f zero8() {
  F32x8 z;
#pragma unroll
  for (int i = 0; i < 8; ++i) z.e[i] = 0.0f;
  return z.v;
}

__device__ __forceinline__ v8f wmma_bf16(v16bf a, v16bf b, v8f c) {
  return __builtin_amdgcn_wmma_f32_16x16x32_bf16(false, a, false, b, (short)0, c, false, false);
}

// B fragment (32x16 bf16) from LDS column-major data.
// lane = (n = lane&15 -> column col0 + n*colstep ; half = lane>>4 -> K += 16)
// 16 contiguous bf16 per lane -> two 16B LDS loads.
__device__ __forceinline__ v16bf ldsB(const __bf16* base, int stride, int col0,
                                      int colstep, int koff, int lane) {
  const int n = lane & 15, hh = lane >> 4;
  const __bf16* p = base + (size_t)(col0 + n * colstep) * stride + koff + hh * 16;
  BF16x16 r;
  r.u[0] = *(const uint4*)(p);
  r.u[1] = *(const uint4*)(p + 8);
  return r.v;
}

// A fragment (16x32 bf16) loaded from a global f32 weight matrix (row-major, ld cols).
// A k order used here (contiguous, matches ldsB): lane m = lane&15 -> row ; element e -> K = e + 16*(lane>>4).
__device__ __forceinline__ v16bf gA(const float* __restrict__ W, int ld, int row,
                                    int kbase, int lane, int kmax) {
  const int hh = lane >> 4;
  const float* wr = W + (size_t)row * ld;
  BF16x16 r;
#pragma unroll
  for (int e = 0; e < 16; ++e) {
    int K = kbase + e + 16 * hh;
    float f = (K < kmax) ? wr[K] : 0.0f;
    r.e[e] = (__bf16)f;
  }
  return r.v;
}

// Pack 8 f32 -> 8 contiguous bf16 (16B) and store with one ds_store_b128.
__device__ __forceinline__ void st8bf(__bf16* dst, v8f val) {
  F32x8 f; f.v = val;
  BF16x8 b;
#pragma unroll
  for (int e = 0; e < 8; ++e) b.e[e] = (__bf16)f.e[e];
  *(uint4*)dst = b.u;
}

__global__ __launch_bounds__(256) void fused_pointnet_kernel(
    const float* __restrict__ feats, const float* __restrict__ ppfs,
    const float* __restrict__ w0, const float* __restrict__ b0,
    const float* __restrict__ w1, const float* __restrict__ b1,
    const float* __restrict__ w2, const float* __restrict__ b2,
    const float* __restrict__ w3, const float* __restrict__ b3,
    const float* __restrict__ ww1, const float* __restrict__ ww2,
    const float* __restrict__ bw2,
    const float* __restrict__ wo, const float* __restrict__ bo,
    const float* __restrict__ p1, const float* __restrict__ p2,
    float* __restrict__ out) {
  extern __shared__ __align__(16) char smem[];
  __bf16* s_arena = (__bf16*)(smem + OFF_ARENA);  // [256][136] feats -> xn
  __bf16* s_pp    = (__bf16*)(smem + OFF_PP);     // [256][72]  ppfs -> pt
  __bf16* s_xfs   = (__bf16*)(smem + OFF_XFS);    // [16][1032]
  float*  s_x1    = (float*) (smem + OFF_X1);     // [64][16]
  float*  s_h     = (float*) (smem + OFF_H);      // [16][8]
  float*  s_part  = (float*) (smem + OFF_PART);   // [8][32][8]
  float*  s_wsm   = (float*) (smem + OFF_WSM);    // [8][16][16]
  __bf16* s_om    = (__bf16*)(smem + OFF_OM);     // [16][72]
  float*  s_id    = (float*) (smem + OFF_ID);     // [128][16]

  const int tid  = threadIdx.x;
  const int wv   = tid >> 5;
  const int lane = tid & 31;
  const int hh   = lane >> 4;
  const int ln   = lane & 15;
  const int bb   = blockIdx.x >> 10;      // batch
  const int grp  = blockIdx.x & 1023;     // 16-point group
  const int n0   = grp * GROUP;

  // ---------------- P0: stage feats + ppfs into column-major bf16 LDS ----------------
  {
    const float* fb = feats + (size_t)bb * CIN * ROWSTRIDE + (size_t)n0 * KD;
    const int cofs = tid >> 6;          // 4 channels per iter
    const int col4 = (tid & 63) * 4;    // 64 threads cover 256 cols
    for (int cc = 0; cc < CIN; cc += 4) {
      const int c = cc + cofs;
      const float4 f = *(const float4*)(fb + (size_t)c * ROWSTRIDE + col4);
      s_arena[(col4 + 0) * AR_STR + c] = (__bf16)f.x;
      s_arena[(col4 + 1) * AR_STR + c] = (__bf16)f.y;
      s_arena[(col4 + 2) * AR_STR + c] = (__bf16)f.z;
      s_arena[(col4 + 3) * AR_STR + c] = (__bf16)f.w;
    }
    const float* pb = ppfs + (size_t)bb * PTC * ROWSTRIDE + (size_t)n0 * KD;
    const int col = tid;  // 256 threads -> 256 cols
#pragma unroll
    for (int c = 0; c < PTC; ++c)
      s_pp[col * PP_STR + c] = (__bf16)pb[(size_t)c * ROWSTRIDE + col];
#pragma unroll
    for (int c = PTC; c < 32; ++c) s_pp[col * PP_STR + c] = (__bf16)0.0f;
  }
  __syncthreads();

  // ---------------- P1: conv0 (128x128 @ 128x256), relu+bias, xn -> arena -------------
  v8f xn[16];
  {
    v16bf a[4];
#pragma unroll
    for (int q = 0; q < 4; ++q) a[q] = gA(w0, CIN, wv * 16 + ln, q * 32, lane, CIN);
    float bias[8];
#pragma unroll
    for (int v = 0; v < 8; ++v) bias[v] = b0[wv * 16 + v + 8 * hh];
#pragma unroll
    for (int t = 0; t < 16; ++t) {
      v8f acc = zero8();
#pragma unroll
      for (int q = 0; q < 4; ++q)
        acc = wmma_bf16(a[q], ldsB(s_arena, AR_STR, t * 16, 1, q * 32, lane), acc);
      F32x8 r; r.v = acc;
#pragma unroll
      for (int v = 0; v < 8; ++v) {
        float x = r.e[v] + bias[v];
        r.e[v] = x > 0.0f ? x : 0.0f;
      }
      xn[t] = r.v;
    }
  }
  __syncthreads();  // all feats reads done -> safe to overwrite arena
#pragma unroll
  for (int t = 0; t < 16; ++t) {
    const int col = t * 16 + ln;
    st8bf(s_arena + (size_t)col * AR_STR + wv * 16 + 8 * hh, xn[t]);
    if (ln == 0) {  // col%16==0 -> identity (xn[...,0]) kept in f32
      F32x8 r; r.v = xn[t];
#pragma unroll
      for (int v = 0; v < 8; ++v) s_id[(wv * 16 + v + 8 * hh) * 16 + t] = r.e[v];
    }
  }
  __syncthreads();

  // ---------------- P2: pt1 (waves 0-3) and x1 = conv1 @ xn[:,k=0] (waves 4-7) --------
  if (wv < 4) {
    v16bf a = gA(p1, PTC, wv * 16 + ln, 0, lane, PTC);
#pragma unroll
    for (int t = 0; t < 16; ++t) {
      v8f acc = wmma_bf16(a, ldsB(s_pp, PP_STR, t * 16, 1, 0, lane), zero8());
      F32x8 r; r.v = acc;
#pragma unroll
      for (int v = 0; v < 8; ++v) r.e[v] = r.e[v] > 0.0f ? r.e[v] : 0.0f;  // relu
      xn[t] = r.v;  // reuse dead regs: pt tile t
    }
  } else {
    const int j = wv - 4;
    v16bf a[4];
#pragma unroll
    for (int q = 0; q < 4; ++q) a[q] = gA(w1, CIN, j * 16 + ln, q * 32, lane, CIN);
    v8f acc = zero8();
#pragma unroll
    for (int q = 0; q < 4; ++q)
      acc = wmma_bf16(a[q], ldsB(s_arena, AR_STR, 0, 16, q * 32, lane), acc);
    F32x8 r; r.v = acc;
#pragma unroll
    for (int v = 0; v < 8; ++v) r.e[v] += b1[j * 16 + v + 8 * hh];
    xn[0] = r.v;
  }
  __syncthreads();
  if (wv < 4) {  // pt overwrites ppfs staging (consumed above)
#pragma unroll
    for (int t = 0; t < 16; ++t)
      st8bf(s_pp + (size_t)(t * 16 + ln) * PP_STR + wv * 16 + 8 * hh, xn[t]);
  } else {
    const int j = wv - 4;
    F32x8 r; r.v = xn[0];
#pragma unroll
    for (int v = 0; v < 8; ++v) s_x1[(j * 16 + v + 8 * hh) * 16 + ln] = r.e[v];
  }
  __syncthreads();

  // ---------------- P3: per-wave 2 points: ptf, x2->xfs, x3 --------------------------
  v8f ptf[4][2], x3a[4][2];
#pragma unroll
  for (int j = 0; j < 4; ++j) {
    v16bf a0 = gA(p2, MIDC, j * 16 + ln, 0, lane, MIDC);
    v16bf a1 = gA(p2, MIDC, j * 16 + ln, 32, lane, MIDC);
#pragma unroll
    for (int pp = 0; pp < 2; ++pp) {
      const int p = 2 * wv + pp;
      v8f acc = wmma_bf16(a0, ldsB(s_pp, PP_STR, p * 16, 1, 0, lane), zero8());
      acc = wmma_bf16(a1, ldsB(s_pp, PP_STR, p * 16, 1, 32, lane), acc);
      ptf[j][pp] = acc;
    }
  }
#pragma unroll
  for (int j = 0; j < 4; ++j) {  // x2 and xfs = x1 - x2 + ptf
    v16bf a[4];
#pragma unroll
    for (int q = 0; q < 4; ++q) a[q] = gA(w2, CIN, j * 16 + ln, q * 32, lane, CIN);
    float bias[8];
#pragma unroll
    for (int v = 0; v < 8; ++v) bias[v] = b2[j * 16 + v + 8 * hh];
#pragma unroll
    for (int pp = 0; pp < 2; ++pp) {
      const int p = 2 * wv + pp;
      v8f acc = zero8();
#pragma unroll
      for (int q = 0; q < 4; ++q)
        acc = wmma_bf16(a[q], ldsB(s_arena, AR_STR, p * 16, 1, q * 32, lane), acc);
      F32x8 r, pf; r.v = acc; pf.v = ptf[j][pp];
#pragma unroll
      for (int v = 0; v < 8; ++v) {
        const int c = 16 * j + v + 8 * hh;
        float xf = s_x1[c * 16 + p] - (r.e[v] + bias[v]) + pf.e[v];
        s_xfs[(size_t)p * XFS_STR + c * 16 + ln] = (__bf16)xf;  // flat = c*16+k
      }
    }
  }
#pragma unroll
  for (int j = 0; j < 4; ++j) {  // x3 = conv3(xn) + b3 + ptf (kept in regs)
    v16bf a[4];
#pragma unroll
    for (int q = 0; q < 4; ++q) a[q] = gA(w3, CIN, j * 16 + ln, q * 32, lane, CIN);
    float bias[8];
#pragma unroll
    for (int v = 0; v < 8; ++v) bias[v] = b3[j * 16 + v + 8 * hh];
#pragma unroll
    for (int pp = 0; pp < 2; ++pp) {
      const int p = 2 * wv + pp;
      v8f acc = zero8();
#pragma unroll
      for (int q = 0; q < 4; ++q)
        acc = wmma_bf16(a[q], ldsB(s_arena, AR_STR, p * 16, 1, q * 32, lane), acc);
      F32x8 r, pf; r.v = acc; pf.v = ptf[j][pp];
#pragma unroll
      for (int v = 0; v < 8; ++v) r.e[v] += bias[v] + pf.e[v];
      x3a[j][pp] = r.v;
    }
  }
  __syncthreads();

  // ---------------- P4: h = relu(convw1 @ xfs), K=1024 split across 8 waves ----------
  {
    v8f acc = zero8();
#pragma unroll
    for (int i = 0; i < 4; ++i) {
      const int q = 4 * wv + i;
      v16bf a = gA(ww1, 1024, ln, q * 32, lane, (ln < 8) ? 1024 : 0);
      acc = wmma_bf16(a, ldsB(s_xfs, XFS_STR, 0, 1, q * 32, lane), acc);
    }
    F32x8 r; r.v = acc;
#pragma unroll
    for (int v = 0; v < 8; ++v) s_part[(wv * 32 + lane) * 8 + v] = r.e[v];
  }
  __syncthreads();
  {
    const int li = tid >> 3, vi = tid & 7;
    float s = 0.0f;
#pragma unroll
    for (int ww = 0; ww < 8; ++ww) s += s_part[(ww * 32 + li) * 8 + vi];
    s = s > 0.0f ? s : 0.0f;
    if (li < 16) s_h[(li & 15) * 8 + vi] = s;  // rows 0..7 valid only
  }
  __syncthreads();

  // ---------------- P5: logits = convw2 @ h + bias; softmax over k=16 ----------------
  {
    v16bf a = gA(ww2, 8, wv * 16 + ln, 0, lane, 8);
    BF16x16 bfr;
#pragma unroll
    for (int e = 0; e < 16; ++e) {
      const int K = e + 16 * hh;
      bfr.e[e] = (K < 8) ? (__bf16)s_h[ln * 8 + K] : (__bf16)0.0f;
    }
    v8f acc = wmma_bf16(a, bfr.v, zero8());
    F32x8 r; r.v = acc;
    float mx = -1e30f;
#pragma unroll
    for (int v = 0; v < 8; ++v) {
      r.e[v] += bw2[wv * 16 + v + 8 * hh];
      mx = fmaxf(mx, r.e[v]);
    }
    mx = fmaxf(mx, __shfl_xor(mx, 16, 32));
    float sum = 0.0f;
#pragma unroll
    for (int v = 0; v < 8; ++v) { r.e[v] = __expf(r.e[v] - mx); sum += r.e[v]; }
    sum += __shfl_xor(sum, 16, 32);
    const float inv = 1.0f / sum;
#pragma unroll
    for (int v = 0; v < 8; ++v)
      s_wsm[(wv * 16 + v + 8 * hh) * 16 + ln] = r.e[v] * inv;  // [g][k][p]
  }
  __syncthreads();

  // ---------------- P6: out_mid = relu(sum_k wsoft[c%8,k] * x3[c,k]) -----------------
#pragma unroll
  for (int j = 0; j < 4; ++j) {
#pragma unroll
    for (int pp = 0; pp < 2; ++pp) {
      const int p = 2 * wv + pp;
      F32x8 r; r.v = x3a[j][pp];
#pragma unroll
      for (int v = 0; v < 8; ++v) {  // c = 16j+v+8h -> group = v
        float x = r.e[v] * s_wsm[(v * 16 + ln) * 16 + p];
        x += __shfl_xor(x, 1, 32);
        x += __shfl_xor(x, 2, 32);
        x += __shfl_xor(x, 4, 32);
        x += __shfl_xor(x, 8, 32);
        r.e[v] = x > 0.0f ? x : 0.0f;
      }
      if (ln == 0) st8bf(s_om + (size_t)p * PP_STR + 16 * j + 8 * hh, r.v);
    }
  }
  __syncthreads();

  // ---------------- P7: out = convout(out_mid) + bias + identity ---------------------
  {
    v16bf a0 = gA(wo, MIDC, wv * 16 + ln, 0, lane, MIDC);
    v16bf a1 = gA(wo, MIDC, wv * 16 + ln, 32, lane, MIDC);
    v8f acc = wmma_bf16(a0, ldsB(s_om, PP_STR, 0, 1, 0, lane), zero8());
    acc = wmma_bf16(a1, ldsB(s_om, PP_STR, 0, 1, 32, lane), acc);
    F32x8 r; r.v = acc;
    const int p = ln;
    float* ob = out + (size_t)bb * OUTC * NPTS + n0 + p;
#pragma unroll
    for (int v = 0; v < 8; ++v) {
      const int o = wv * 16 + v + 8 * hh;
      ob[(size_t)o * NPTS] = r.e[v] + bo[o] + s_id[o * 16 + p];
    }
  }
}

extern "C" void kernel_launch(void* const* d_in, const int* in_sizes, int n_in,
                              void* d_out, int out_size, void* d_ws, size_t ws_size,
                              hipStream_t stream) {
  (void)in_sizes; (void)n_in; (void)d_ws; (void)ws_size; (void)out_size;
  const float* feats = (const float*)d_in[0];
  const float* pp    = (const float*)d_in[1];
  const float* w0    = (const float*)d_in[2];
  const float* b0    = (const float*)d_in[3];
  const float* w1    = (const float*)d_in[4];
  const float* b1    = (const float*)d_in[5];
  const float* w2    = (const float*)d_in[6];
  const float* b2    = (const float*)d_in[7];
  const float* w3    = (const float*)d_in[8];
  const float* b3    = (const float*)d_in[9];
  const float* ww1   = (const float*)d_in[10];
  const float* ww2   = (const float*)d_in[11];
  const float* bw2   = (const float*)d_in[12];
  const float* wo    = (const float*)d_in[13];
  const float* bo    = (const float*)d_in[14];
  const float* p1    = (const float*)d_in[15];
  const float* p2    = (const float*)d_in[16];

  dim3 grid(4 * (NPTS / GROUP));  // 4096 blocks: (batch, 16-point group)
  dim3 block(256);                // 8 wave32
  fused_pointnet_kernel<<<grid, block, SMEM_TOTAL, stream>>>(
      feats, pp, w0, b0, w1, b1, w2, b2, w3, b3, ww1, ww2, bw2, wo, bo, p1, p2,
      (float*)d_out);
}
